// STGATConvLayer_45930380263927
// MI455X (gfx1250) — matrix operand order repaired and checked by reading, not compiled, not run on hardware
//
#include <hip/hip_runtime.h>
#include <hip/hip_bf16.h>

typedef __attribute__((ext_vector_type(16))) _Float16 v16h;
typedef __attribute__((ext_vector_type(8)))  _Float16 v8h;
typedef __attribute__((ext_vector_type(8)))  float    v8f;

#define IN_DIM   128
#define OUT_DIM  64
#define NEG_SLOPE 0.2f

// ---------------------------------------------------------------------------
// K0: init emax = -inf, denom = 0, out = 0
// ---------------------------------------------------------------------------
__global__ __launch_bounds__(256) void gat_init(float* __restrict__ emax,
                                                float* __restrict__ denom,
                                                float* __restrict__ out,
                                                int N, int totalOut) {
    int i = blockIdx.x * 256 + threadIdx.x;
    if (i < totalOut) out[i] = 0.0f;
    if (i < N) {
        emax[i]  = -__builtin_inff();
        denom[i] = 0.0f;
    }
}

// ---------------------------------------------------------------------------
// K1: h = X @ W via v_wmma_f32_16x16x32_f16, fused with el/er computation.
// Block = 256 threads = 8 waves; block tile = 32 rows x 64 cols (2x4 wave
// tiles of 16x16). X tile staged row-major, W staged TRANSPOSED (Wt[col][k])
// in LDS so both A and B fragments are contiguous 16B-aligned LDS vectors.
// D is staged through LDS for coalesced float4 stores, and el/er are reduced
// from the LDS tile (no separate pass over h).
// ---------------------------------------------------------------------------
__global__ __launch_bounds__(256) void gat_gemm(const float* __restrict__ X,
                                                const float* __restrict__ W,
                                                const float* __restrict__ a_l,
                                                const float* __restrict__ a_r,
                                                float* __restrict__ h,
                                                float* __restrict__ el,
                                                float* __restrict__ er,
                                                int N) {
    __shared__ alignas(16) _Float16 Xs[32][IN_DIM + 8];       // 8704 B
    __shared__ alignas(16) _Float16 Wt[OUT_DIM][IN_DIM + 8];  // 17408 B
    __shared__ alignas(16) float    Hs[32][OUT_DIM + 4];      // 8704 B

    const int tid      = threadIdx.x;
    const int lane     = tid & 31;
    const int wave     = tid >> 5;
    const int blockRow = blockIdx.x * 32;

    // --- stage W (128x64 f32, row-major) -> Wt[col][k] f16 (transposed)
    #pragma unroll
    for (int i = 0; i < 8; ++i) {
        int idx = i * 256 + tid;                  // float4 index in [0, 2048)
        int row = idx >> 4;                       // k index (64 floats per row)
        int col = (idx & 15) << 2;
        float4 v = *(const float4*)(W + (size_t)row * OUT_DIM + col);
        Wt[col + 0][row] = (_Float16)v.x;
        Wt[col + 1][row] = (_Float16)v.y;
        Wt[col + 2][row] = (_Float16)v.z;
        Wt[col + 3][row] = (_Float16)v.w;
    }

    // --- stage X tile (32x128 f32 -> f16)
    #pragma unroll
    for (int i = 0; i < 4; ++i) {
        int idx  = i * 256 + tid;                 // float4 index in [0, 1024)
        int row  = idx >> 5;                      // 32 f4 per row (128 floats)
        int col  = (idx & 31) << 2;
        int grow = blockRow + row;
        if (grow >= N) grow = N - 1;              // clamp; EXEC stays full
        float4 v = *(const float4*)(X + (size_t)grow * IN_DIM + col);
        Xs[row][col + 0] = (_Float16)v.x;
        Xs[row][col + 1] = (_Float16)v.y;
        Xs[row][col + 2] = (_Float16)v.z;
        Xs[row][col + 3] = (_Float16)v.w;
    }
    __syncthreads();

    const int rowTile = wave >> 2;                // 0..1
    const int colTile = wave & 3;                 // 0..3
    const int rowBase = rowTile * 16;
    const int colBase = colTile * 16;
    const int mn = lane & 15;                     // row (A) / col (B) index
    const int hi = lane >> 4;                     // 0: lanes 0-15, 1: lanes 16-31

    const _Float16* arow = &Xs[rowBase + mn][0];
    const _Float16* brow = &Wt[colBase + mn][0];

    v8f acc = {};

    #pragma unroll
    for (int kb = 0; kb < IN_DIM; kb += 32) {
        // A 16x32 f16 fragment: lane (m=mn, half=hi) needs
        //   K = kb + 8*hi + {0..7}  and  K = kb + 16 + 8*hi + {0..7}
        v8h a0 = *(const v8h*)(arow + kb + 8 * hi);
        v8h a1 = *(const v8h*)(arow + kb + 16 + 8 * hi);
        v16h a = __builtin_shufflevector(a0, a1, 0, 1, 2, 3, 4, 5, 6, 7,
                                                 8, 9, 10, 11, 12, 13, 14, 15);
        // B 32x16 f16 fragment: lane (n=mn, half=hi) needs
        //   K = kb + 8*hi + {0..7}  and  K = kb + 16 + 8*hi + {0..7}
        v8h b0 = *(const v8h*)(brow + kb + 8 * hi);
        v8h b1 = *(const v8h*)(brow + kb + 16 + 8 * hi);
        v16h b = __builtin_shufflevector(b0, b1, 0, 1, 2, 3, 4, 5, 6, 7,
                                                 8, 9, 10, 11, 12, 13, 14, 15);

        acc = __builtin_amdgcn_wmma_f32_16x16x32_f16(
                  /*neg_a=*/false, a, /*neg_b=*/false, b,
                  /*c_mod=*/(short)0, acc, /*reuse_a=*/false, /*reuse_b=*/false);
    }

    // --- stage D tile into LDS: VGPR v -> local row (rowBase + 8*hi + v)
    const int lrow0 = rowBase + hi * 8;
    #pragma unroll
    for (int v = 0; v < 8; ++v) {
        Hs[lrow0 + v][colBase + mn] = acc[v];
    }
    __syncthreads();

    // --- coalesced writeback of h: 512 float4 / 256 threads = 2 each
    #pragma unroll
    for (int i = 0; i < 2; ++i) {
        int idx  = i * 256 + tid;                 // float4 index in [0, 512)
        int row  = idx >> 4;                      // 16 f4 per row (64 floats)
        int col  = (idx & 15) << 2;
        int grow = blockRow + row;
        if (grow < N) {
            *(float4*)(h + (size_t)grow * OUT_DIM + col) = *(const float4*)&Hs[row][col];
        }
    }

    // --- fused el/er: 8 lanes per row, partial dot + width-8 shfl reduction
    {
        int row = tid >> 3;                       // 0..31
        int seg = tid & 7;                        // 8-col segment
        float sl = 0.0f, sr = 0.0f;
        #pragma unroll
        for (int j = 0; j < 8; ++j) {
            int c = seg * 8 + j;
            float hv = Hs[row][c];
            sl += hv * a_l[c];
            sr += hv * a_r[c];
        }
        #pragma unroll
        for (int off = 4; off > 0; off >>= 1) {
            sl += __shfl_down(sl, off, 8);
            sr += __shfl_down(sr, off, 8);
        }
        int grow = blockRow + row;
        if (seg == 0 && grow < N) {
            el[grow] = sl;
            er[grow] = sr;
        }
    }
}

// ---------------------------------------------------------------------------
// K2: per-edge logits + segment max via monotonic bit-trick float atomicMax
// ---------------------------------------------------------------------------
__device__ __forceinline__ void atomicMaxF(float* addr, float v) {
    if (v >= 0.0f) atomicMax((int*)addr, __float_as_int(v));
    else           atomicMin((unsigned int*)addr, __float_as_uint(v));
}

__global__ __launch_bounds__(256) void gat_edge_logits(const int* __restrict__ src,
                                                       const int* __restrict__ dst,
                                                       const float* __restrict__ el,
                                                       const float* __restrict__ er,
                                                       float* __restrict__ logits,
                                                       float* __restrict__ emax,
                                                       int E) {
    int e = blockIdx.x * 256 + threadIdx.x;
    if (e >= E) return;
    int s = src[e];
    int d = dst[e];
    float x = el[s] + er[d];
    x = (x > 0.0f) ? x : NEG_SLOPE * x;
    logits[e] = x;
    atomicMaxF(&emax[d], x);
}

// ---------------------------------------------------------------------------
// K3: per-edge w = exp(logit - emax[dst]); denom[dst]+=w;
//     out[dst,:] += w * h[src,:]   (one wave per edge, float2 per lane)
// ---------------------------------------------------------------------------
__global__ __launch_bounds__(256) void gat_scatter(const int* __restrict__ src,
                                                   const int* __restrict__ dst,
                                                   const float* __restrict__ logits,
                                                   const float* __restrict__ emax,
                                                   const float* __restrict__ h,
                                                   float* __restrict__ denom,
                                                   float* __restrict__ out,
                                                   int E) {
    int wave = threadIdx.x >> 5;
    int lane = threadIdx.x & 31;
    int e    = blockIdx.x * 8 + wave;
    if (e >= E) return;

    int s = src[e];
    int d = dst[e];
    float w = __expf(logits[e] - emax[d]);
    if (lane == 0) atomicAdd(&denom[d], w);

    float2 hv = *(const float2*)(h + (size_t)s * OUT_DIM + 2 * lane);
    float* op = out + (size_t)d * OUT_DIM + 2 * lane;
    atomicAdd(op + 0, w * hv.x);
    atomicAdd(op + 1, w * hv.y);
}

// ---------------------------------------------------------------------------
// K4: out[n,:] /= max(denom[n], 1e-16)
// ---------------------------------------------------------------------------
__global__ __launch_bounds__(256) void gat_norm(float* __restrict__ out,
                                                const float* __restrict__ denom,
                                                int total) {
    int i = blockIdx.x * 256 + threadIdx.x;
    if (i >= total) return;
    float dnm = denom[i >> 6];   // OUT_DIM == 64
    out[i] = out[i] / fmaxf(dnm, 1e-16f);
}

// ---------------------------------------------------------------------------
extern "C" void kernel_launch(void* const* d_in, const int* in_sizes, int n_in,
                              void* d_out, int out_size, void* d_ws, size_t ws_size,
                              hipStream_t stream) {
    const float* X   = (const float*)d_in[0];      // [N, 128]
    const int*   ei  = (const int*)d_in[1];        // [2, E] int32
    const float* W   = (const float*)d_in[2];      // [128, 64]
    const float* a_l = (const float*)d_in[3];      // [64]
    const float* a_r = (const float*)d_in[4];      // [64]
    float*       out = (float*)d_out;              // [N, 64]

    const int N = in_sizes[0] / IN_DIM;
    const int E = in_sizes[1] / 2;

    // workspace carve-up (floats): h | el | er | emax | denom | logits
    float* h      = (float*)d_ws;
    float* el     = h + (size_t)N * OUT_DIM;
    float* er     = el + N;
    float* emax   = er + N;
    float* denom  = emax + N;
    float* logits = denom + N;

    const int* src = ei;
    const int* dst = ei + E;

    const int totalOut = N * OUT_DIM;
    dim3 blk(256);

    gat_init<<<(totalOut + 255) / 256, blk, 0, stream>>>(emax, denom, out, N, totalOut);
    gat_gemm<<<(N + 31) / 32, blk, 0, stream>>>(X, W, a_l, a_r, h, el, er, N);
    gat_edge_logits<<<(E + 255) / 256, blk, 0, stream>>>(src, dst, el, er, logits, emax, E);
    gat_scatter<<<(E + 7) / 8, blk, 0, stream>>>(src, dst, logits, emax, h, denom, out, E);
    gat_norm<<<(totalOut + 255) / 256, blk, 0, stream>>>(out, denom, totalOut);
}
